// CATSI_2602750181978
// MI455X (gfx1250) — compile-verified, matrix operand order checked by reading, not run on hardware
//
#include <hip/hip_runtime.h>
#include <hip/hip_bf16.h>
#include <math.h>

#define B_ 128
#define T_ 128
#define V_ 128
#define H_ 64
#define CH_ 32
#define FH_ 32
#define BT_ (B_*T_)
#define BTV_ (B_*T_*V_)

typedef __attribute__((ext_vector_type(16))) _Float16 v16h_t;
typedef __attribute__((ext_vector_type(8)))  _Float16 v8h_t;
typedef __attribute__((ext_vector_type(8)))  float    v8f_t;

__device__ __forceinline__ float sigf(float x) { return 1.0f / (1.0f + expf(-x)); }

// ---------------------------------------------------------------------------
// Utility kernels
// ---------------------------------------------------------------------------
__global__ void zero_scalars_kernel(float* p, int n) {
    int i = blockIdx.x * blockDim.x + threadIdx.x;
    if (i < n) p[i] = 0.0f;
}

__global__ void cvt_f16_kernel(const float* __restrict__ src, _Float16* __restrict__ dst, int n) {
    int i = blockIdx.x * blockDim.x + threadIdx.x;
    if (i < n) dst[i] = (_Float16)src[i];
}

// feat_W (V,FH,V) with fm applied: zero v==o column
__global__ void cvt_featW_kernel(const float* __restrict__ src, _Float16* __restrict__ dst) {
    int i = blockIdx.x * blockDim.x + threadIdx.x;
    if (i < V_ * FH_ * V_) {
        int v = i % V_;
        int o = i / (FH_ * V_);
        dst[i] = (v == o) ? (_Float16)0.0f : (_Float16)src[i];
    }
}

// ---------------------------------------------------------------------------
// Stats / normalization / decay pass.  One block per batch row, thread = v.
// Emits nv (f32), pad, stats, and the f16 GEMM A-matrices.
// ---------------------------------------------------------------------------
__global__ void stats_kernel(const float* __restrict__ values,
                             const float* __restrict__ masks,
                             const float* __restrict__ deltas,
                             const int*   __restrict__ lengths,
                             const float* __restrict__ minv,
                             const float* __restrict__ maxv,
                             const float* __restrict__ decW,
                             const float* __restrict__ decb,
                             float* __restrict__ nv,
                             float* __restrict__ pad,
                             float* __restrict__ stats,
                             _Float16* __restrict__ xcomp16,
                             _Float16* __restrict__ gin16,
                             _Float16* __restrict__ fuseA16,
                             float* __restrict__ masksum)
{
    int b = blockIdx.x;
    int v = threadIdx.x;
    int len = lengths[b];
    float mn  = minv[b * V_ + v];
    float mmn = maxv[b * V_ + v] - mn;

    float msum = 0.0f, vsum = 0.0f;
    for (int t = 0; t < T_; ++t) {
        size_t idx = ((size_t)b * T_ + t) * V_ + v;
        msum += masks[idx];
        vsum += values[idx];
    }
    float mean = vsum / msum;
    float var = 0.0f;
    for (int t = 0; t < T_; ++t) {
        size_t idx = ((size_t)b * T_ + t) * V_ + v;
        float d = values[idx] - mean;
        var += d * d;
    }
    var /= (msum - 1.0f);
    float sd   = sqrtf(var);
    float miss = 1.0f - msum / (float)len;

    const int SW = 3 * V_ + 1;
    stats[b * SW + 1 + v]           = mean;
    stats[b * SW + 1 + V_ + v]      = sd;
    stats[b * SW + 1 + 2 * V_ + v]  = miss;
    if (v == 0) stats[b * SW] = (float)len;
    atomicAdd(masksum, msum);

    float nmean = (mean - mn) / mmn;
    float ddiag = decW[v * V_ + v];
    float db    = decb[v];
    float prev  = 0.0f;
    for (int t = 0; t < T_; ++t) {
        size_t idx = ((size_t)b * T_ + t) * V_ + v;
        float val = values[idx], m = masks[idx], de = deltas[idx];
        float nvt = (val - mn) / mmn;
        float pm  = (t < len) ? 1.0f : 0.0f;
        if (v == 0) pad[b * T_ + t] = pm;
        float xp = (t == 0) ? nvt : prev;
        float g  = expf(-fmaxf(de * ddiag + db, 0.0f));
        float xd = g * xp + (1.0f - g) * nmean;
        float xc = (m * nvt + (1.0f - m) * xd) * pm;
        nv[idx] = nvt;
        xcomp16[idx] = (_Float16)xc;
        size_t r = (size_t)b * T_ + t;
        gin16[r * 256 + v]       = (_Float16)xc;
        gin16[r * 256 + V_ + v]  = (_Float16)de;
        fuseA16[r * 256 + v]      = (_Float16)g;
        fuseA16[r * 256 + V_ + v] = (_Float16)m;
        prev = nvt;
    }
}

// ---------------------------------------------------------------------------
// Generic WMMA GEMM: C(M,N)f32 = act(A(M,K)f16 @ W(N,K)f16^T + bias)
// wave32: one wave computes a 16x32 output tile (2 accumulators, A reused).
// K % 32 == 0, N % 32 == 0, M % 16 == 0.
// ---------------------------------------------------------------------------
__global__ void wmma_gemm_bias_act(const _Float16* __restrict__ A,
                                   const _Float16* __restrict__ W,
                                   const float* __restrict__ bias,
                                   float* __restrict__ C,
                                   int M, int N, int K, int relu)
{
    const int lane  = threadIdx.x & 31;
    const int gwave = blockIdx.x * (blockDim.x >> 5) + (threadIdx.x >> 5);
    const int tn2   = N >> 5;
    const int total = (M >> 4) * tn2;
    if (gwave >= total) return;
    const int tm   = gwave / tn2;
    const int tc   = gwave % tn2;
    const int r    = lane & 15;
    const int half = lane >> 4;

    const _Float16* arow  = A + (size_t)(tm * 16 + r) * K + half * 8;
    const _Float16* brow0 = W + (size_t)(tc * 32 + r) * K + half * 8;
    const _Float16* brow1 = W + (size_t)(tc * 32 + 16 + r) * K + half * 8;

    v8f_t acc0 = {};
    v8f_t acc1 = {};
    for (int k = 0; k < K; k += 32) {
        if (k + 72 <= K) __builtin_prefetch(arow + k + 64, 0, 1);
        v8h_t a0 = *(const v8h_t*)(arow + k);
        v8h_t a1 = *(const v8h_t*)(arow + k + 16);
        v8h_t p0 = *(const v8h_t*)(brow0 + k);
        v8h_t p1 = *(const v8h_t*)(brow0 + k + 16);
        v8h_t q0 = *(const v8h_t*)(brow1 + k);
        v8h_t q1 = *(const v8h_t*)(brow1 + k + 16);
        v16h_t av, bv, cv;
#pragma unroll
        for (int i = 0; i < 8; ++i) {
            av[i] = a0[i]; av[8 + i] = a1[i];
            bv[i] = p0[i]; bv[8 + i] = p1[i];
            cv[i] = q0[i]; cv[8 + i] = q1[i];
        }
        acc0 = __builtin_amdgcn_wmma_f32_16x16x32_f16(false, av, false, bv, (short)0, acc0, false, false);
        acc1 = __builtin_amdgcn_wmma_f32_16x16x32_f16(false, av, false, cv, (short)0, acc1, false, false);
    }
    const int n0 = tc * 32 + r;
    const float bn0 = bias ? bias[n0] : 0.0f;
    const float bn1 = bias ? bias[n0 + 16] : 0.0f;
#pragma unroll
    for (int m8 = 0; m8 < 8; ++m8) {
        int m = tm * 16 + half * 8 + m8;
        float v0 = acc0[m8] + bn0;
        float v1 = acc1[m8] + bn1;
        if (relu) { v0 = fmaxf(v0, 0.0f); v1 = fmaxf(v1, 0.0f); }
        C[(size_t)m * N + n0]      = v0;
        C[(size_t)m * N + n0 + 16] = v1;
    }
}

// ---------------------------------------------------------------------------
// Context MLP: stats(B,385) -> relu -> (B,32).  Tiny, scalar fp32.
// ---------------------------------------------------------------------------
__global__ void ctx_mlp_kernel(const float* __restrict__ stats,
                               const float* __restrict__ W1, const float* __restrict__ b1,
                               const float* __restrict__ W2, const float* __restrict__ b2,
                               float* __restrict__ ctx_mlp)
{
    __shared__ float s[3 * V_ + 1];
    __shared__ float h1[2 * CH_];
    const int SW = 3 * V_ + 1;
    int b = blockIdx.x, t = threadIdx.x;
    for (int i = t; i < SW; i += 64) s[i] = stats[b * SW + i];
    __syncthreads();
    float acc = b1[t];
    for (int k = 0; k < SW; ++k) acc += s[k] * W1[t * SW + k];
    h1[t] = fmaxf(acc, 0.0f);
    __syncthreads();
    if (t < CH_) {
        float a = b2[t];
        for (int k = 0; k < 2 * CH_; ++k) a += h1[k] * W2[t * 2 * CH_ + k];
        ctx_mlp[b * CH_ + t] = a;
    }
}

// ---------------------------------------------------------------------------
// GRU scan: one block per batch row, 96 threads.  gi precomputed by WMMA GEMM.
// ---------------------------------------------------------------------------
__global__ void gru_scan_kernel(const float* __restrict__ gi,   // (B,T,96)
                                const float* __restrict__ Whh,  // (96,32)
                                const float* __restrict__ bhh,  // (96)
                                const float* __restrict__ pad,  // (B,T)
                                float* __restrict__ ctx_rnn)    // (B,32)
{
    __shared__ float h[CH_], hn[CH_], gh[3 * CH_];
    int b = blockIdx.x, tid = threadIdx.x;
    if (tid < CH_) { h[tid] = 0.0f; hn[tid] = 0.0f; }
    __syncthreads();
    for (int t = 0; t < T_; ++t) {
        float a = bhh[tid];
        const float* row = Whh + tid * CH_;
        for (int k = 0; k < CH_; ++k) a += h[k] * row[k];
        gh[tid] = a;
        __syncthreads();
        if (tid < CH_) {
            const float* gir = gi + ((size_t)b * T_ + t) * (3 * CH_);
            float r  = sigf(gir[tid]            + gh[tid]);
            float z  = sigf(gir[CH_ + tid]      + gh[CH_ + tid]);
            float n  = tanhf(gir[2 * CH_ + tid] + r * gh[2 * CH_ + tid]);
            float h2 = (1.0f - z) * n + z * h[tid];
            float pm = pad[b * T_ + t];
            hn[tid] = pm * h2 + (1.0f - pm) * hn[tid];
            h[tid]  = h2;
        }
        __syncthreads();
    }
    if (tid < CH_) ctx_rnn[b * CH_ + tid] = hn[tid];
}

// ---------------------------------------------------------------------------
// context_vec = [ctx_mlp, ctx_rnn]; h0 = cv @ initW^T + initb; c0 = tanh(h0)
// ---------------------------------------------------------------------------
__global__ void init_kernel(const float* __restrict__ ctx_mlp,
                            const float* __restrict__ ctx_rnn,
                            const float* __restrict__ initW,  // (128,64)
                            const float* __restrict__ initb,  // (128)
                            float* __restrict__ context_vec,  // (B,64)
                            float* __restrict__ h0,           // (B,128)
                            float* __restrict__ c0)           // (B,128)
{
    __shared__ float cv[2 * CH_];
    int b = blockIdx.x, t = threadIdx.x;
    if (t < CH_)           cv[t] = ctx_mlp[b * CH_ + t];
    else if (t < 2 * CH_)  cv[t] = ctx_rnn[b * CH_ + (t - CH_)];
    __syncthreads();
    if (t < 2 * CH_) context_vec[b * 2 * CH_ + t] = cv[t];
    float a = initb[t];
    for (int k = 0; k < 2 * CH_; ++k) a += cv[k] * initW[t * 2 * CH_ + k];
    h0[b * 2 * H_ + t] = a;
    c0[b * 2 * H_ + t] = tanhf(a);
}

// ---------------------------------------------------------------------------
// Build LSTM input matrix f16: (BT, 320) = [x_comp, masks, context_vec[b]]
// ---------------------------------------------------------------------------
__global__ void build_inputs_kernel(const _Float16* __restrict__ xcomp16,
                                    const float* __restrict__ masks,
                                    const float* __restrict__ context_vec,
                                    _Float16* __restrict__ inputs16)
{
    int bt = blockIdx.x;
    int b  = bt / T_;
    int i  = threadIdx.x;
    _Float16 val;
    if (i < V_)            val = xcomp16[(size_t)bt * V_ + i];
    else if (i < 2 * V_)   val = (_Float16)masks[(size_t)bt * V_ + (i - V_)];
    else                   val = (_Float16)context_vec[b * 2 * CH_ + (i - 2 * V_)];
    inputs16[(size_t)bt * 320 + i] = val;
}

// ---------------------------------------------------------------------------
// Bidirectional LSTM scan.  grid (B, 2), 256 threads; gi preprojected.
// Writes hid16 (BT, 128) f16 directly for the rec WMMA GEMM.
// ---------------------------------------------------------------------------
__global__ void lstm_scan_kernel(const float* __restrict__ gif,  // (B,T,256)
                                 const float* __restrict__ gib,  // (B,T,256)
                                 const float* __restrict__ Whhf, const float* __restrict__ bhhf,
                                 const float* __restrict__ Whhb, const float* __restrict__ bhhb,
                                 const float* __restrict__ h0, const float* __restrict__ c0,
                                 _Float16* __restrict__ hid16)   // (B,T,128)
{
    __shared__ float h[H_];
    __shared__ float g[4 * H_];
    int b = blockIdx.x, dir = blockIdx.y, tid = threadIdx.x;
    const float* gx  = dir ? gib  : gif;
    const float* Whh = dir ? Whhb : Whhf;
    const float* bhh = dir ? bhhb : bhhf;
    float c = 0.0f;
    if (tid < H_) {
        h[tid] = h0[b * 2 * H_ + dir * H_ + tid];
        c      = c0[b * 2 * H_ + dir * H_ + tid];
    }
    __syncthreads();

    if (dir == 0) {
        for (int t = 0; t < T_ - 1; ++t) {
            if (tid < H_) hid16[((size_t)b * T_ + t) * 2 * H_ + tid] = (_Float16)h[tid];
            float a = bhh[tid];
            const float* row = Whh + tid * H_;
            for (int k = 0; k < H_; ++k) a += h[k] * row[k];
            a += gx[((size_t)b * T_ + t) * 4 * H_ + tid];
            g[tid] = a;
            __syncthreads();
            if (tid < H_) {
                float i_ = sigf(g[tid]);
                float f_ = sigf(g[H_ + tid]);
                float gg = tanhf(g[2 * H_ + tid]);
                float o_ = sigf(g[3 * H_ + tid]);
                c = f_ * c + i_ * gg;
                h[tid] = o_ * tanhf(c);
            }
            __syncthreads();
        }
        if (tid < H_) hid16[((size_t)b * T_ + (T_ - 1)) * 2 * H_ + tid] = (_Float16)h[tid];
    } else {
        if (tid < H_) hid16[((size_t)b * T_ + (T_ - 1)) * 2 * H_ + H_ + tid] = (_Float16)h[tid];
        for (int t = T_ - 1; t >= 1; --t) {
            float a = bhh[tid];
            const float* row = Whh + tid * H_;
            for (int k = 0; k < H_; ++k) a += h[k] * row[k];
            a += gx[((size_t)b * T_ + t) * 4 * H_ + tid];
            g[tid] = a;
            __syncthreads();
            if (tid < H_) {
                float i_ = sigf(g[tid]);
                float f_ = sigf(g[H_ + tid]);
                float gg = tanhf(g[2 * H_ + tid]);
                float o_ = sigf(g[3 * H_ + tid]);
                c = f_ * c + i_ * gg;
                h[tid] = o_ * tanhf(c);
            }
            __syncthreads();
            if (tid < H_) hid16[((size_t)b * T_ + (t - 1)) * 2 * H_ + H_ + tid] = (_Float16)h[tid];
        }
    }
}

// ---------------------------------------------------------------------------
// Fused feat path: for one output feature o and a 32-row tile:
//   z = relu(x_comp @ featW[o]^T + feat_b[o])      (WMMA, K=128)
//   z = relu(z @ regW1^T + reg_b1)                 (WMMA, K=32, via LDS transpose)
//   feat_imp[:, o] = z @ regW2 + reg_b2
// grid: (BT/32, V), 128 threads (4 waves = 2x2 tiles of 16x16).
// ---------------------------------------------------------------------------
__global__ void feat_kernel(const _Float16* __restrict__ xcomp16,  // (BT,128)
                            const _Float16* __restrict__ featW16,  // (V,32,128) masked
                            const float* __restrict__ feat_b,      // (V,32)
                            const _Float16* __restrict__ regW1_16, // (32,32)
                            const float* __restrict__ reg_b1,
                            const float* __restrict__ reg_W2,
                            const float* __restrict__ reg_b2,
                            float* __restrict__ feat_imp)          // (BT,V)
{
    __shared__ _Float16 z1[32 * 40];
    __shared__ float    z2[32 * 36];
    int o    = blockIdx.y;
    int row0 = blockIdx.x * 32;
    int tid  = threadIdx.x;
    int lane = tid & 31, wid = tid >> 5;
    int wm = wid >> 1, wn = wid & 1;
    int r = lane & 15, half = lane >> 4;

    // GEMM1
    const _Float16* arow = xcomp16 + (size_t)(row0 + wm * 16 + r) * V_ + half * 8;
    const _Float16* brow = featW16 + (size_t)o * FH_ * V_ + (size_t)(wn * 16 + r) * V_ + half * 8;
    v8f_t acc = {};
    for (int k = 0; k < V_; k += 32) {
        v8h_t a0 = *(const v8h_t*)(arow + k);
        v8h_t a1 = *(const v8h_t*)(arow + k + 16);
        v8h_t b0 = *(const v8h_t*)(brow + k);
        v8h_t b1 = *(const v8h_t*)(brow + k + 16);
        v16h_t av, bv;
#pragma unroll
        for (int i = 0; i < 8; ++i) { av[i] = a0[i]; av[8 + i] = a1[i]; bv[i] = b0[i]; bv[8 + i] = b1[i]; }
        acc = __builtin_amdgcn_wmma_f32_16x16x32_f16(false, av, false, bv, (short)0, acc, false, false);
    }
    float bn = feat_b[o * FH_ + wn * 16 + r];
#pragma unroll
    for (int m8 = 0; m8 < 8; ++m8) {
        float v = fmaxf(acc[m8] + bn, 0.0f);
        z1[(wm * 16 + half * 8 + m8) * 40 + wn * 16 + r] = (_Float16)v;
    }
    __syncthreads();

    // GEMM2 (K=32, one wmma): A from LDS z1, B = regW1
    v16h_t a2, b2;
    const _Float16* w1row = regW1_16 + (wn * 16 + r) * FH_;
#pragma unroll
    for (int i = 0; i < 8; ++i) {
        a2[i]     = z1[(wm * 16 + r) * 40 + half * 8 + i];
        a2[8 + i] = z1[(wm * 16 + r) * 40 + 16 + half * 8 + i];
        b2[i]     = w1row[half * 8 + i];
        b2[8 + i] = w1row[16 + half * 8 + i];
    }
    v8f_t acc2 = {};
    acc2 = __builtin_amdgcn_wmma_f32_16x16x32_f16(false, a2, false, b2, (short)0, acc2, false, false);
    float b1n = reg_b1[wn * 16 + r];
#pragma unroll
    for (int m8 = 0; m8 < 8; ++m8) {
        z2[(wm * 16 + half * 8 + m8) * 36 + wn * 16 + r] = fmaxf(acc2[m8] + b1n, 0.0f);
    }
    __syncthreads();

    if (tid < 32) {
        float s = reg_b2[0];
        for (int hh = 0; hh < FH_; ++hh) s += z2[tid * 36 + hh] * reg_W2[hh];
        feat_imp[(size_t)(row0 + tid) * V_ + o] = s;
    }
}

// ---------------------------------------------------------------------------
// Fusion, loss reduction, rescale outputs.
// ---------------------------------------------------------------------------
__global__ void final_kernel(const float* __restrict__ nv,
                             const float* __restrict__ masks,
                             const float* __restrict__ pad,
                             const float* __restrict__ minv,
                             const float* __restrict__ maxv,
                             const float* __restrict__ rnn_imp,
                             const float* __restrict__ feat_imp,
                             const float* __restrict__ logits,
                             float* __restrict__ out_final,
                             float* __restrict__ out_feat,
                             float* __restrict__ out_rnn,
                             float* __restrict__ loss_acc)
{
    __shared__ float red[256];
    int tid = threadIdx.x;
    size_t idx = (size_t)blockIdx.x * 256 + tid;
    int v = (int)(idx & (V_ - 1));
    size_t bt = idx >> 7;
    int t = (int)(bt & (T_ - 1));
    int b = (int)(bt >> 7);

    float m   = masks[idx];
    float nvv = nv[idx];
    float beta = sigf(logits[idx]);
    float fi = feat_imp[idx];
    float ri = rnn_imp[idx];
    float fu = beta * fi + (1.0f - beta) * ri;
    float fin = m * nvv + (1.0f - m) * fu;
    float d1 = (ri - nvv) * m, d2 = (fi - nvv) * m, d3 = (fu - nvv) * m;
    red[tid] = d1 * d1 + d2 * d2 + d3 * d3;

    float pm  = pad[b * T_ + t];
    float mn  = minv[b * V_ + v];
    float mmn = maxv[b * V_ + v] - mn;
    out_final[idx] = (pm > 0.5f) ? fin * mmn + mn : 0.0f;
    out_feat[idx]  = (pm > 0.5f) ? fi * mmn + mn : 0.0f;
    out_rnn[idx]   = (pm > 0.5f) ? ri * mmn + mn : 0.0f;

    __syncthreads();
    for (int s = 128; s > 0; s >>= 1) {
        if (tid < s) red[tid] += red[tid + s];
        __syncthreads();
    }
    if (tid == 0) atomicAdd(loss_acc, red[0]);
}

__global__ void finalize_loss_kernel(const float* __restrict__ loss_acc,
                                     const float* __restrict__ masksum,
                                     float* __restrict__ out0)
{
    if (threadIdx.x == 0 && blockIdx.x == 0) out0[0] = loss_acc[0] / masksum[0];
}

// ---------------------------------------------------------------------------
// Host launcher
// ---------------------------------------------------------------------------
static inline int divup(int a, int b) { return (a + b - 1) / b; }

extern "C" void kernel_launch(void* const* d_in, const int* in_sizes, int n_in,
                              void* d_out, int out_size, void* d_ws, size_t ws_size,
                              hipStream_t stream)
{
    (void)in_sizes; (void)n_in; (void)out_size; (void)ws_size;

    const float* values   = (const float*)d_in[0];
    const float* masks    = (const float*)d_in[1];
    const float* deltas   = (const float*)d_in[2];
    const int*   lengths  = (const int*)  d_in[3];
    const float* min_vals = (const float*)d_in[4];
    const float* max_vals = (const float*)d_in[5];
    const float* dec_W    = (const float*)d_in[6];
    const float* dec_b    = (const float*)d_in[7];
    const float* ctx_W1   = (const float*)d_in[8];
    const float* ctx_b1   = (const float*)d_in[9];
    const float* ctx_W2   = (const float*)d_in[10];
    const float* ctx_b2   = (const float*)d_in[11];
    const float* gru_Wih  = (const float*)d_in[12];
    const float* gru_Whh  = (const float*)d_in[13];
    const float* gru_bih  = (const float*)d_in[14];
    const float* gru_bhh  = (const float*)d_in[15];
    const float* init_W   = (const float*)d_in[16];
    const float* init_b   = (const float*)d_in[17];
    const float* lf_Wih   = (const float*)d_in[18];
    const float* lf_Whh   = (const float*)d_in[19];
    const float* lf_bih   = (const float*)d_in[20];
    const float* lf_bhh   = (const float*)d_in[21];
    const float* lb_Wih   = (const float*)d_in[22];
    const float* lb_Whh   = (const float*)d_in[23];
    const float* lb_bih   = (const float*)d_in[24];
    const float* lb_bhh   = (const float*)d_in[25];
    const float* rec_W    = (const float*)d_in[26];
    const float* rec_b    = (const float*)d_in[27];
    const float* feat_W   = (const float*)d_in[28];
    const float* feat_b   = (const float*)d_in[29];
    const float* reg_W1   = (const float*)d_in[30];
    const float* reg_b1   = (const float*)d_in[31];
    const float* reg_W2   = (const float*)d_in[32];
    const float* reg_b2   = (const float*)d_in[33];
    const float* fuse_W   = (const float*)d_in[34];
    const float* fuse_b   = (const float*)d_in[35];

    float* out = (float*)d_out;
    float* out_final = out + 1;
    float* out_feat  = out + 1 + (size_t)BTV_;
    float* out_rnn   = out + 1 + 2 * (size_t)BTV_;

    char* ws = (char*)d_ws;
    size_t off = 0;
    auto alloc = [&](size_t bytes) -> char* {
        char* p = ws + off;
        off = (off + bytes + 255) & ~(size_t)255;
        return p;
    };

    float*    nv        = (float*)   alloc((size_t)BTV_ * 4);
    _Float16* xcomp16   = (_Float16*)alloc((size_t)BTV_ * 2);
    _Float16* gin16     = (_Float16*)alloc((size_t)BT_ * 256 * 2);
    _Float16* fuseA16   = (_Float16*)alloc((size_t)BT_ * 256 * 2);
    _Float16* inputs16  = (_Float16*)alloc((size_t)BT_ * 320 * 2);
    float*    pad       = (float*)   alloc((size_t)B_ * T_ * 4);
    float*    stats     = (float*)   alloc((size_t)B_ * (3 * V_ + 1) * 4);
    float*    ctx_mlp   = (float*)   alloc((size_t)B_ * CH_ * 4);
    float*    ctx_rnn   = (float*)   alloc((size_t)B_ * CH_ * 4);
    float*    ctxvec    = (float*)   alloc((size_t)B_ * 2 * CH_ * 4);
    float*    h0        = (float*)   alloc((size_t)B_ * 2 * H_ * 4);
    float*    c0        = (float*)   alloc((size_t)B_ * 2 * H_ * 4);
    float*    gi_gru    = (float*)   alloc((size_t)BT_ * 96 * 4);
    float*    gif       = (float*)   alloc((size_t)BT_ * 256 * 4);
    float*    gib       = (float*)   alloc((size_t)BT_ * 256 * 4);
    _Float16* hid16     = (_Float16*)alloc((size_t)BT_ * 128 * 2);
    float*    rnn_imp   = (float*)   alloc((size_t)BTV_ * 4);
    float*    feat_imp  = (float*)   alloc((size_t)BTV_ * 4);
    float*    logits    = (float*)   alloc((size_t)BTV_ * 4);
    _Float16* lfWih16   = (_Float16*)alloc((size_t)256 * 320 * 2);
    _Float16* lbWih16   = (_Float16*)alloc((size_t)256 * 320 * 2);
    _Float16* gruWih16  = (_Float16*)alloc((size_t)96 * 256 * 2);
    _Float16* recW16    = (_Float16*)alloc((size_t)128 * 128 * 2);
    _Float16* fuseW16   = (_Float16*)alloc((size_t)128 * 256 * 2);
    _Float16* featW16   = (_Float16*)alloc((size_t)V_ * FH_ * V_ * 2);
    _Float16* regW1_16  = (_Float16*)alloc((size_t)FH_ * FH_ * 2);
    float*    scalars   = (float*)   alloc(2 * 4);  // [loss_acc, masksum]

    // 0) zero accumulators
    zero_scalars_kernel<<<1, 32, 0, stream>>>(scalars, 2);

    // 1) weight conversions
    cvt_f16_kernel<<<divup(256 * 320, 256), 256, 0, stream>>>(lf_Wih, lfWih16, 256 * 320);
    cvt_f16_kernel<<<divup(256 * 320, 256), 256, 0, stream>>>(lb_Wih, lbWih16, 256 * 320);
    cvt_f16_kernel<<<divup(96 * 256, 256), 256, 0, stream>>>(gru_Wih, gruWih16, 96 * 256);
    cvt_f16_kernel<<<divup(128 * 128, 256), 256, 0, stream>>>(rec_W, recW16, 128 * 128);
    cvt_f16_kernel<<<divup(128 * 256, 256), 256, 0, stream>>>(fuse_W, fuseW16, 128 * 256);
    cvt_f16_kernel<<<divup(FH_ * FH_, 256), 256, 0, stream>>>(reg_W1, regW1_16, FH_ * FH_);
    cvt_featW_kernel<<<divup(V_ * FH_ * V_, 256), 256, 0, stream>>>(feat_W, featW16);

    // 2) stats / normalization / decay (writes nv, pad, stats, f16 A-matrices)
    stats_kernel<<<B_, V_, 0, stream>>>(values, masks, deltas, lengths, min_vals, max_vals,
                                        dec_W, dec_b, nv, pad, stats,
                                        xcomp16, gin16, fuseA16, scalars + 1);

    // 3) context MLP
    ctx_mlp_kernel<<<B_, 64, 0, stream>>>(stats, ctx_W1, ctx_b1, ctx_W2, ctx_b2, ctx_mlp);

    // 4) GRU input projection (WMMA) + scan
    {
        int waves = (BT_ / 16) * (96 / 32);
        wmma_gemm_bias_act<<<divup(waves, 8), 256, 0, stream>>>(gin16, gruWih16, gru_bih,
                                                               gi_gru, BT_, 96, 256, 0);
    }
    gru_scan_kernel<<<B_, 96, 0, stream>>>(gi_gru, gru_Whh, gru_bhh, pad, ctx_rnn);

    // 5) init state + LSTM input matrix
    init_kernel<<<B_, 128, 0, stream>>>(ctx_mlp, ctx_rnn, init_W, init_b, ctxvec, h0, c0);
    build_inputs_kernel<<<BT_, 320, 0, stream>>>(xcomp16, masks, ctxvec, inputs16);

    // 6) LSTM input projections (WMMA) + bidirectional scan
    {
        int waves = (BT_ / 16) * (256 / 32);
        wmma_gemm_bias_act<<<divup(waves, 8), 256, 0, stream>>>(inputs16, lfWih16, lf_bih,
                                                               gif, BT_, 256, 320, 0);
        wmma_gemm_bias_act<<<divup(waves, 8), 256, 0, stream>>>(inputs16, lbWih16, lb_bih,
                                                               gib, BT_, 256, 320, 0);
    }
    lstm_scan_kernel<<<dim3(B_, 2), 256, 0, stream>>>(gif, gib, lf_Whh, lf_bhh, lb_Whh, lb_bhh,
                                                      h0, c0, hid16);

    // 7) rec projection + fuse gate logits (WMMA)
    {
        int waves = (BT_ / 16) * (128 / 32);
        wmma_gemm_bias_act<<<divup(waves, 8), 256, 0, stream>>>(hid16, recW16, rec_b,
                                                               rnn_imp, BT_, 128, 128, 0);
        wmma_gemm_bias_act<<<divup(waves, 8), 256, 0, stream>>>(fuseA16, fuseW16, fuse_b,
                                                               logits, BT_, 128, 256, 0);
    }

    // 8) fused feature-regression path (WMMA chain)
    feat_kernel<<<dim3(BT_ / 32, V_), 128, 0, stream>>>(xcomp16, featW16, feat_b,
                                                        regW1_16, reg_b1, reg_W2, reg_b2,
                                                        feat_imp);

    // 9) fusion + loss + rescale
    final_kernel<<<BTV_ / 256, 256, 0, stream>>>(nv, masks, pad, min_vals, max_vals,
                                                 rnn_imp, feat_imp, logits,
                                                 out_final, out_feat, out_rnn, scalars);
    finalize_loss_kernel<<<1, 32, 0, stream>>>(scalars, scalars + 1, out);
}
